// RHC_25417616457992
// MI455X (gfx1250) — compile-verified, matrix operand order checked by reading, not run on hardware
//
#include <hip/hip_runtime.h>

// ---------------------------------------------------------------------------
// Adaptive-halting LSTM, persistent-kernel formulation for MI455X (gfx1250).
//   T=400, B=256, V=128, H=512, C=64.
//   grid = 16 WGs x 256 threads; each WG owns 16 batch rows and runs the whole
//   400-step recurrence with all state in LDS/registers. GEMMs via bf16 WMMA
//   with f32 accumulation. Weights pre-packed to WMMA B-fragment layout once.
// ---------------------------------------------------------------------------

#define T_STEPS 400
#define BATCH   256
#define VFEAT   128
#define HID     512
#define CLS     64
#define ACT_K   768           // [x:128 | ybar:64 | h:512 | yhat:64]
#define NT_G    128           // gate N-tiles (2048/16)
#define KT_G    22            // gate K-tiles (704/32)
#define KT_O    16            // W_out K-tiles (512/32)
#define KT_C    18            // W_ctrl K-tiles (576/32)
#define WR_BLK  (KT_G * NT_G) // 2816
#define WO_BLK  (KT_O * 4)    // 64
#define WC_BLK  (KT_C * 4)    // 72
#define NBLK    (WR_BLK + WO_BLK + WC_BLK) // 2952 fragment blocks, 1KB each
#define NWG     (BATCH / 16)  // 16

typedef __attribute__((ext_vector_type(16))) __bf16       v16bf;
typedef __attribute__((ext_vector_type(8)))  float        v8f;
typedef __attribute__((ext_vector_type(8)))  unsigned int v8u;

static __device__ __forceinline__ unsigned int bfbits(float f) {
  unsigned int u = __builtin_bit_cast(unsigned int, f);
  u += 0x7FFFu + ((u >> 16) & 1u);          // round-to-nearest-even
  return u >> 16;
}
static __device__ __forceinline__ unsigned int pack2(float lo, float hi) {
  return bfbits(lo) | (bfbits(hi) << 16);
}
static __device__ __forceinline__ float sigm(float x) {
  return 1.0f / (1.0f + __expf(-x));
}
static __device__ __forceinline__ v16bf asbf(v8u u) {
  return __builtin_bit_cast(v16bf, u);
}
static __device__ __forceinline__ v8f wmma_bf16(v16bf a, v16bf b, v8f c) {
  // D(16x16 f32) = A(16x32 bf16) x B(32x16 bf16) + C
  return __builtin_amdgcn_wmma_f32_16x16x32_bf16(false, a, false, b,
                                                 (short)0, c, false, false);
}
// Build an A-fragment from the bf16 activation buffer in LDS.
// 16-bit A 16x32 layout: lane holds row m=lane%16; vgpr v holds K-pair at
// kk = (v/4)*16 + (lane/16)*8 + (v%4)*2.
static __device__ __forceinline__ v16bf load_a(const unsigned int* actu,
                                               int m, int half, int kbase) {
  v8u a;
#pragma unroll
  for (int v = 0; v < 8; ++v) {
    const int kk = kbase + (v >> 2) * 16 + half * 8 + (v & 3) * 2;
    a[v] = actu[m * (ACT_K / 2) + (kk >> 1)];
  }
  return asbf(a);
}

// ---------------------------------------------------------------------------
// Pack [W_ih | W_hh] (2048x704), W_out (64x512), W_ctrl[:, :576] (64x576) into
// bf16 WMMA B-fragments: per 32x16 (KxN) block, lane L stores its 8 packed
// dwords contiguously (32 B) so the main kernel loads each fragment as b128x2.
// ---------------------------------------------------------------------------
__global__ void rhc_pack(const float* __restrict__ W_ih,
                         const float* __restrict__ W_hh,
                         const float* __restrict__ W_out,
                         const float* __restrict__ W_ctrl,
                         v8u* __restrict__ ws) {
  const int gid  = blockIdx.x;
  const int lane = threadIdx.x;
  const int half = lane >> 4;
  const int ln   = lane & 15;
  v8u o;
  if (gid < WR_BLK) {                       // recurrent weights, K order [x|ybar|h]
    const int kt = gid >> 7, nt = gid & 127;
    const int n  = nt * 16 + ln;            // gate column 0..2047
#pragma unroll
    for (int v = 0; v < 8; ++v) {
      const int kk = kt * 32 + (v >> 2) * 16 + half * 8 + (v & 3) * 2;
      float lo, hi;
      if (kk < 192) { lo = W_ih[n * 192 + kk];       hi = W_ih[n * 192 + kk + 1]; }
      else          { lo = W_hh[n * 512 + kk - 192]; hi = W_hh[n * 512 + kk - 191]; }
      o[v] = pack2(lo, hi);
    }
  } else if (gid < WR_BLK + WO_BLK) {       // W_out, K order [h]
    const int b  = gid - WR_BLK;
    const int kt = b >> 2, nt = b & 3;
    const int n  = nt * 16 + ln;
#pragma unroll
    for (int v = 0; v < 8; ++v) {
      const int kk = kt * 32 + (v >> 2) * 16 + half * 8 + (v & 3) * 2;
      o[v] = pack2(W_out[n * 512 + kk], W_out[n * 512 + kk + 1]);
    }
  } else {                                  // W_ctrl[:, :576], K order [h|yhat]
    const int b  = gid - WR_BLK - WO_BLK;
    const int kt = b >> 2, nt = b & 3;
    const int n  = nt * 16 + ln;
#pragma unroll
    for (int v = 0; v < 8; ++v) {
      const int kk = kt * 32 + (v >> 2) * 16 + half * 8 + (v & 3) * 2;
      o[v] = pack2(W_ctrl[n * 577 + kk], W_ctrl[n * 577 + kk + 1]);
    }
  }
  ws[(size_t)gid * 32 + lane] = o;
}

// ---------------------------------------------------------------------------
// Persistent recurrent kernel: one WG per 16 batch rows.
// ---------------------------------------------------------------------------
__global__ __launch_bounds__(256) void rhc_lstm(
    const float* __restrict__ X, const float* __restrict__ noise,
    const float* __restrict__ b_ih, const float* __restrict__ b_hh,
    const float* __restrict__ b_out, const float* __restrict__ b_ctrl,
    const float* __restrict__ W_ctrl, const int* __restrict__ epoch_p,
    const v8u* __restrict__ wr, const v8u* __restrict__ wo,
    const v8u* __restrict__ wc,
    float* __restrict__ y_out, float* __restrict__ partials) {
  __shared__ __align__(16) unsigned short act[16 * ACT_K]; // bf16 [x|ybar|h|yhat]
  __shared__ float cst[16 * HID];                          // f32 cell state
  __shared__ float red[128];

  const int tid  = threadIdx.x;
  const int w    = tid >> 5;                // wave 0..7
  const int lane = tid & 31;
  const int half = lane >> 4;
  const int ln   = lane & 15;
  const int rowbase = blockIdx.x * 16;

  const float eps = __expf(-7.0f * (float)(*epoch_p) / 99.0f); // linspace(0,7,100)[epoch]

  for (int i = tid; i < 16 * ACT_K; i += 256) act[i] = 0;
  for (int i = tid; i < 16 * HID;  i += 256) cst[i] = 0.0f;

  // Hoist per-lane constants: gate bias per owned N-column, controller consts.
  float bias_g[16];
#pragma unroll
  for (int j = 0; j < 16; ++j) {
    const int nt = (j >> 2) * 32 + w * 4 + (j & 3);   // wave w owns h-range w*64..
    const int n  = nt * 16 + ln;
    bias_g[j] = b_ih[n] + b_hh[n];
  }
  const int cidx = w * 16 + ln;                        // class idx (waves 0..3)
  float bo = 0.f, bc = 0.f, wct = 0.f;
  if (w < 4) { bo = b_out[cidx]; bc = b_ctrl[cidx]; wct = W_ctrl[cidx * 577 + 576]; }

  float preds_r[8], halt_r[8], yv[8];
#pragma unroll
  for (int r = 0; r < 8; ++r) { preds_r[r] = 0.f; halt_r[r] = -1.f; yv[r] = 0.f; }

  const unsigned int* actu = (const unsigned int*)act;
  __syncthreads();

  for (int t = 0; t < T_STEPS; ++t) {
    // ---- phase 0: stream x_t into LDS (f32 -> bf16), prefetch x_{t+1} ----
    {
      const float* xp = X + (size_t)t * BATCH * VFEAT + (size_t)rowbase * VFEAT;
      const int e = tid * 8;
      const int m = e >> 7, vf = e & 127;
#pragma unroll
      for (int i = 0; i < 8; ++i)
        act[m * ACT_K + vf + i] = (unsigned short)bfbits(xp[m * VFEAT + vf + i]);
      if (t + 1 < T_STEPS)
        __builtin_prefetch(xp + (size_t)BATCH * VFEAT + e, 0, 0);
    }
    __syncthreads();

    // ---- phase 1: gates = [x|ybar|h] @ [W_ih|W_hh]^T + b  (16 x 2048) ----
    v8f acc[16];
#pragma unroll
    for (int j = 0; j < 16; ++j) {
      const float b = bias_g[j];
#pragma unroll
      for (int r = 0; r < 8; ++r) acc[j][r] = b;
    }
    for (int kt = 0; kt < KT_G; ++kt) {
      const v16bf a = load_a(actu, ln, half, kt * 32);
      const v8u* wrow = wr + (size_t)(kt * NT_G) * 32;
#pragma unroll
      for (int j = 0; j < 16; ++j) {
        const int nt = (j >> 2) * 32 + w * 4 + (j & 3);
        acc[j] = wmma_bf16(a, asbf(wrow[nt * 32 + lane]), acc[j]);
      }
    }
    __syncthreads();   // all waves finished reading old h / ybar from act

    // ---- phase 1b: LSTM pointwise; wave w has i,f,g,o for its h-range ----
#pragma unroll
    for (int q = 0; q < 4; ++q) {
#pragma unroll
      for (int r = 0; r < 8; ++r) {
        const int m = half * 8 + r;
        const int hidx = w * 64 + q * 16 + ln;
        const float iv = sigm(acc[q][r]);
        const float fv = sigm(acc[4 + q][r]);
        const float gv = tanhf(acc[8 + q][r]);
        const float ov = sigm(acc[12 + q][r]);
        const float cn = fv * cst[m * HID + hidx] + iv * gv;
        cst[m * HID + hidx] = cn;
        act[m * ACT_K + 192 + hidx] = (unsigned short)bfbits(ov * tanhf(cn));
      }
    }
    __syncthreads();

    // ---- phase 2: y_hat = sigmoid(h @ W_out^T + b_out)  (waves 0..3) ----
    if (w < 4) {
      v8f ay;
#pragma unroll
      for (int r = 0; r < 8; ++r) ay[r] = bo;
      for (int kt = 0; kt < KT_O; ++kt) {
        const v16bf a = load_a(actu, ln, half, 192 + kt * 32);
        ay = wmma_bf16(a, asbf(wo[(size_t)(kt * 4 + w) * 32 + lane]), ay);
      }
#pragma unroll
      for (int r = 0; r < 8; ++r) {
        yv[r] = sigm(ay[r]);
        const int m = half * 8 + r;
        act[m * ACT_K + 704 + cidx] = (unsigned short)bfbits(yv[r]);
      }
    }
    __syncthreads();

    // ---- phase 3: controller + halting state updates (waves 0..3) ----
    if (w < 4) {
      v8f ac;
      const float cinit = bc + (float)t * wct;   // bias + time column
#pragma unroll
      for (int r = 0; r < 8; ++r) ac[r] = cinit;
      for (int kt = 0; kt < KT_C; ++kt) {        // K = [h(512) | yhat(64)]
        const v16bf a = load_a(actu, ln, half, 192 + kt * 32);
        ac = wmma_bf16(a, asbf(wc[(size_t)(kt * 4 + w) * 32 + lane]), ac);
      }
      const float* np = noise + (size_t)t * BATCH * CLS + (size_t)rowbase * CLS;
#pragma unroll
      for (int r = 0; r < 8; ++r) {
        const int m = half * 8 + r;
        float p = sigm(ac[r]);
        p = (1.0f - eps) * p + eps * 0.05f;
        const bool at = np[m * CLS + cidx] < p;
        if (at && preds_r[r] == 0.0f) preds_r[r] = yv[r];
        if (at && act[m * ACT_K + 128 + cidx] == 0)
          act[m * ACT_K + 128 + cidx] = 0x3F80;  // ybar := bf16(1.0)
        if (at && halt_r[r] == -1.0f) halt_r[r] = (float)t;
      }
    }
    // next iteration's phase-0 writes touch only the x region; its barrier
    // orders this phase's ybar writes before the next gates GEMM.
  }
  __syncthreads();

  // ---- epilogue: final predictions + deterministic halting-point partial ----
  float lsum = 0.0f;
  if (w < 4) {
#pragma unroll
    for (int r = 0; r < 8; ++r) {
      const int m = half * 8 + r;
      const float yf = (preds_r[r] == 0.0f) ? yv[r] : preds_r[r];
      y_out[(size_t)(rowbase + m) * CLS + cidx] = yf;
      const float hv = (halt_r[r] == -1.0f) ? (float)(T_STEPS - 1) : halt_r[r];
      lsum += 1.0f + hv;
    }
    red[tid] = lsum;
  }
  __syncthreads();
  if (tid == 0) {
    float s = 0.0f;
    for (int i = 0; i < 128; ++i) s += red[i];
    partials[blockIdx.x] = s;
  }
}

__global__ void rhc_reduce(const float* __restrict__ partials,
                           float* __restrict__ out_scalar) {
  if (blockIdx.x == 0 && threadIdx.x == 0) {
    float s = 0.0f;
    for (int i = 0; i < NWG; ++i) s += partials[i];
    out_scalar[0] = s / ((float)(BATCH * CLS) * (float)(T_STEPS + 1));
  }
}

extern "C" void kernel_launch(void* const* d_in, const int* in_sizes, int n_in,
                              void* d_out, int out_size, void* d_ws, size_t ws_size,
                              hipStream_t stream) {
  (void)in_sizes; (void)n_in; (void)out_size; (void)ws_size;
  const float* X      = (const float*)d_in[0];
  const float* noise  = (const float*)d_in[1];
  const float* W_ih   = (const float*)d_in[2];
  const float* W_hh   = (const float*)d_in[3];
  const float* b_ih   = (const float*)d_in[4];
  const float* b_hh   = (const float*)d_in[5];
  const float* W_out  = (const float*)d_in[6];
  const float* b_out  = (const float*)d_in[7];
  const float* W_ctrl = (const float*)d_in[8];
  const float* b_ctrl = (const float*)d_in[9];
  const int*   epoch  = (const int*)d_in[10];
  float* out = (float*)d_out;

  v8u* ws = (v8u*)d_ws;                       // NBLK * 1KB packed fragments
  float* partials = (float*)((char*)d_ws + (size_t)NBLK * 1024);

  rhc_pack<<<NBLK, 32, 0, stream>>>(W_ih, W_hh, W_out, W_ctrl, ws);
  rhc_lstm<<<NWG, 256, 0, stream>>>(X, noise, b_ih, b_hh, b_out, b_ctrl,
                                    W_ctrl, epoch,
                                    ws,                    // Wr fragments
                                    ws + (size_t)WR_BLK * 32,            // Wo
                                    ws + (size_t)(WR_BLK + WO_BLK) * 32, // Wc
                                    out, partials);
  rhc_reduce<<<1, 32, 0, stream>>>(partials, out + BATCH * CLS);
}
